// SpatialDecoder_652835029744
// MI455X (gfx1250) — matrix operand order repaired and checked by reading, not compiled, not run on hardware
//
#include <hip/hip_runtime.h>

// ---------------------------------------------------------------------------
// Shapes: B=128, N=1024, DM=64, S=2, D_IN=68
// Outputs: read [128,1,1024] (131072 f32) then out [128,128,1024] (16777216 f32)
// Workspace layout (bytes):
//   ws_xin  bf16 [128][64][1024]   @ 0          (16,777,216)
//   ws_adj  bf16 [2][1024][1024]   @ 16777216   ( 4,194,304)
//   ws_hT   bf16 [128][1024][64]   @ 20971520   (16,777,216)
//   ws_wgc  bf16 [64][128]         @ 37748736   (    16,384)
//   ws_wout bf16 [64][128]         @ 37765120   (    16,384)
// total ~37.8 MB
// ---------------------------------------------------------------------------

typedef __bf16 v16bf __attribute__((ext_vector_type(16)));
typedef __bf16 v8bf  __attribute__((ext_vector_type(8)));
typedef float  v8f   __attribute__((ext_vector_type(8)));

static __device__ __forceinline__ unsigned pack2bf(float a, float b) {
  __bf16 x = (__bf16)a, y = (__bf16)b;
  unsigned short ux = __builtin_bit_cast(unsigned short, x);
  unsigned short uy = __builtin_bit_cast(unsigned short, y);
  return ((unsigned)uy << 16) | (unsigned)ux;
}

// A operand (16x32 MxK, bf16). Lane L holds row M=L%16, half hl=L/16.
// element j -> K = k0 + 16*hl + j : 16 contiguous bf16 at p (16B aligned).
static __device__ __forceinline__ v16bf load_A(const __bf16* p) {
  v8bf lo = *(const v8bf*)p;
  v8bf hi = *(const v8bf*)(p + 8);
  v16bf a;
#pragma unroll
  for (int j = 0; j < 8; ++j) { a[j] = lo[j]; a[j + 8] = hi[j]; }
  return a;
}

// B operand (32x16 KxN, bf16). Lane L holds column N=L%16, half hl=L/16.
// element j -> K = k0 + 16*hl + j : 16 contiguous bf16 at p (16B aligned).
static __device__ __forceinline__ v16bf load_B(const __bf16* p) {
  v8bf lo = *(const v8bf*)p;
  v8bf hi = *(const v8bf*)(p + 8);
  v16bf a;
#pragma unroll
  for (int j = 0; j < 8; ++j) { a[j] = lo[j]; a[j + 8] = hi[j]; }
  return a;
}

// ---------------------------------------------------------------------------
// Prep: bf16 weight conversion (tiny)
// ---------------------------------------------------------------------------
__global__ void k_prep_w(const float* __restrict__ W_gc,
                         const float* __restrict__ W_out,
                         __bf16* __restrict__ ws_wgc,
                         __bf16* __restrict__ ws_wout) {
  for (int i = threadIdx.x; i < 64 * 128; i += 256) {
    ws_wgc[i]  = (__bf16)W_gc[i];
    ws_wout[i] = (__bf16)W_out[i];
  }
}

// ---------------------------------------------------------------------------
// Prep: adj -> bf16 (2M elements, 4 per thread, packed 8B stores)
// ---------------------------------------------------------------------------
__global__ void k_prep_adj(const float* __restrict__ adj,
                           __bf16* __restrict__ ws_adj) {
  size_t i = ((size_t)blockIdx.x * 256 + threadIdx.x) * 4;
  float4 v = *(const float4*)(adj + i);
  unsigned lo = pack2bf(v.x, v.y);
  unsigned hi = pack2bf(v.z, v.w);
  *(uint2*)(void*)(ws_adj + i) = make_uint2(lo, hi);
}

// ---------------------------------------------------------------------------
// Prep: x_in = W_in @ cat(x,m,u,h) + b_in  -> bf16 [b][64][n]
// one thread per (b,n); 64 f32 accumulators in registers; W_in via s_loads.
// ---------------------------------------------------------------------------
__global__ void k_prep_xin(const float* __restrict__ x, const float* __restrict__ m,
                           const float* __restrict__ u, const float* __restrict__ h,
                           const float* __restrict__ W_in, const float* __restrict__ b_in,
                           __bf16* __restrict__ ws_xin) {
  int idx = blockIdx.x * 256 + threadIdx.x;   // 0 .. 131071
  int b = idx >> 10, n = idx & 1023;
  float xv = x[(size_t)b * 1024 + n];
  float mv = m[(size_t)b * 1024 + n];
  float u0 = u[((size_t)b * 2 + 0) * 1024 + n];
  float u1 = u[((size_t)b * 2 + 1) * 1024 + n];
  float acc[64];
#pragma unroll
  for (int co = 0; co < 64; ++co) {
    const float* w = W_in + co * 68;
    acc[co] = b_in[co] + w[0] * xv + w[1] * mv + w[2] * u0 + w[3] * u1;
  }
  for (int c = 0; c < 64; ++c) {
    float hv = h[((size_t)b * 64 + c) * 1024 + n];
#pragma unroll
    for (int co = 0; co < 64; ++co) acc[co] += W_in[co * 68 + 4 + c] * hv;
  }
#pragma unroll
  for (int co = 0; co < 64; ++co)
    ws_xin[((size_t)b * 64 + co) * 1024 + n] = (__bf16)acc[co];
}

// ---------------------------------------------------------------------------
// Prep: h -> bf16 transposed [b][n][64] (LDS tile transpose) + f32 copy of h
// into out[:, 64:128, :].  One block per (b, 64-column tile).
// ---------------------------------------------------------------------------
__global__ void k_prep_h(const float* __restrict__ h,
                         __bf16* __restrict__ ws_hT,
                         float* __restrict__ out_full) {
  __shared__ float tile[64][65];
  int b = blockIdx.x >> 4;
  int n0 = (blockIdx.x & 15) * 64;
#pragma unroll
  for (int rep = 0; rep < 16; ++rep) {
    int l = rep * 256 + threadIdx.x;
    int c = l >> 6, nn = l & 63;
    float v = h[((size_t)b * 64 + c) * 1024 + n0 + nn];
    tile[c][nn] = v;
    out_full[((size_t)b * 128 + 64 + c) * 1024 + n0 + nn] = v;   // out = cat(prelu, h)
  }
  __syncthreads();
#pragma unroll
  for (int rep = 0; rep < 16; ++rep) {
    int l = rep * 256 + threadIdx.x;
    int nl = l >> 6, c = l & 63;
    ws_hT[((size_t)b * 1024 + n0 + nl) * 64 + c] = (__bf16)tile[c][nl];
  }
}

// ---------------------------------------------------------------------------
// Main fused kernel: per block = (batch, 64-col tile), 4 waves, wave owns 16 cols.
//   agg[s]  = x_in @ adj[s]^T          (2 x 1024-K dot, 256 WMMA / wave)
//   gc      = W_gc @ agg + b_gc        (16 WMMA)
//   pre     = W_out @ cat(gc,h)+b_out  (16 WMMA), PReLU -> out[:,0:64,:]
//   read    = W_read @ cat(pre,h) + b_read
// ---------------------------------------------------------------------------
__global__ __launch_bounds__(128) void k_main(
    const float* __restrict__ h,
    const float* __restrict__ b_gc, const float* __restrict__ b_out,
    const float* __restrict__ W_read, const float* __restrict__ b_read,
    const float* __restrict__ prelu_w,
    const __bf16* __restrict__ ws_xin, const __bf16* __restrict__ ws_adj,
    const __bf16* __restrict__ ws_hT, const __bf16* __restrict__ ws_wgc,
    const __bf16* __restrict__ ws_wout,
    float* __restrict__ out_read, float* __restrict__ out_full) {
  // per-wave staging slab, row padded to 136 bf16 (272B) to spread LDS banks
  __shared__ __align__(32) __bf16 stage[4][16][136];

  const int tid = threadIdx.x;
  const int wv = tid >> 5;
  const int lane = tid & 31;
  const int n_l = lane & 15;        // column within wave tile
  const int hl = lane >> 4;         // lane half
  const int b = blockIdx.x >> 4;
  const int w0 = ((blockIdx.x & 15) << 6) + (wv << 4);
  const int wcol = w0 + n_l;

  const __bf16* xb   = ws_xin + (size_t)b * 64 * 1024;
  const __bf16* adj0 = ws_adj + (size_t)wcol * 1024;
  const __bf16* adj1 = adj0 + (size_t)1024 * 1024;

  v8f z;
#pragma unroll
  for (int r = 0; r < 8; ++r) z[r] = 0.0f;

  // ---- Stage 1: agg = x_in @ adj^T, both supports ----
  v8f acc0[4], acc1[4];
#pragma unroll
  for (int mt = 0; mt < 4; ++mt) { acc0[mt] = z; acc1[mt] = z; }

  for (int k0 = 0; k0 < 1024; k0 += 32) {
    if (k0 + 32 < 1024) {
      __builtin_prefetch((const void*)(adj0 + k0 + 32), 0, 3);
      __builtin_prefetch((const void*)(adj1 + k0 + 32), 0, 3);
    }
    v16bf B0 = load_B(adj0 + k0 + 16 * hl);
    v16bf B1 = load_B(adj1 + k0 + 16 * hl);
#pragma unroll
    for (int mt = 0; mt < 4; ++mt) {
      v16bf A = load_A(xb + (size_t)(mt * 16 + n_l) * 1024 + k0 + 16 * hl);
      acc0[mt] = __builtin_amdgcn_wmma_f32_16x16x32_bf16(
          false, A, false, B0, (short)0, acc0[mt], false, false);
      acc1[mt] = __builtin_amdgcn_wmma_f32_16x16x32_bf16(
          false, A, false, B1, (short)0, acc1[mt], false, false);
    }
  }

  // spill agg (C layout: M = r + 8*hl) into stage[w][sc], sc = s*64 + c
  __bf16* srow = &stage[wv][n_l][0];
#pragma unroll
  for (int s = 0; s < 2; ++s)
#pragma unroll
    for (int mt = 0; mt < 4; ++mt) {
      v8f a = s ? acc1[mt] : acc0[mt];
#pragma unroll
      for (int r2 = 0; r2 < 4; ++r2) {
        int sc = s * 64 + mt * 16 + 8 * hl + 2 * r2;
        *(unsigned*)(void*)&srow[sc] = pack2bf(a[2 * r2], a[2 * r2 + 1]);
      }
    }

  // ---- Stage 2: gc = W_gc @ agg + b_gc ----
  v8f gcc[4];
#pragma unroll
  for (int mt = 0; mt < 4; ++mt) gcc[mt] = z;
#pragma unroll
  for (int ks = 0; ks < 4; ++ks) {
    int k0 = ks * 32;
    v16bf Bv = load_B(&stage[wv][n_l][k0 + 16 * hl]);
#pragma unroll
    for (int mt = 0; mt < 4; ++mt) {
      v16bf A = load_A(ws_wgc + (mt * 16 + n_l) * 128 + k0 + 16 * hl);
      gcc[mt] = __builtin_amdgcn_wmma_f32_16x16x32_bf16(
          false, A, false, Bv, (short)0, gcc[mt], false, false);
    }
  }
  // bias, re-stage gc into stage[w][0..63] (same wave => DS kept in order)
#pragma unroll
  for (int mt = 0; mt < 4; ++mt)
#pragma unroll
    for (int r2 = 0; r2 < 4; ++r2) {
      int c0 = mt * 16 + 8 * hl + 2 * r2;
      float g0 = gcc[mt][2 * r2]     + b_gc[c0];
      float g1 = gcc[mt][2 * r2 + 1] + b_gc[c0 + 1];
      *(unsigned*)(void*)&srow[c0] = pack2bf(g0, g1);
    }

  // ---- Stage 3: pre = W_out @ cat(gc, h) ----
  v8f oc[4];
#pragma unroll
  for (int mt = 0; mt < 4; ++mt) oc[mt] = z;
  const __bf16* hrow = ws_hT + ((size_t)b * 1024 + wcol) * 64;
#pragma unroll
  for (int ks = 0; ks < 4; ++ks) {
    int k0 = ks * 32;
    v16bf Bv = (k0 < 64) ? load_B(&stage[wv][n_l][k0 + 16 * hl])
                         : load_B(hrow + (k0 - 64) + 16 * hl);
#pragma unroll
    for (int mt = 0; mt < 4; ++mt) {
      v16bf A = load_A(ws_wout + (mt * 16 + n_l) * 128 + k0 + 16 * hl);
      oc[mt] = __builtin_amdgcn_wmma_f32_16x16x32_bf16(
          false, A, false, Bv, (short)0, oc[mt], false, false);
    }
  }

  // ---- Epilogue: bias + PReLU + store out[:,0:64,:] + readout dot ----
  float pw = prelu_w[0];
  float rp = 0.0f;
#pragma unroll
  for (int mt = 0; mt < 4; ++mt)
#pragma unroll
    for (int r = 0; r < 8; ++r) {
      int c = mt * 16 + 8 * hl + r;
      float v = oc[mt][r] + b_out[c];
      v = (v >= 0.0f) ? v : pw * v;
      out_full[((size_t)b * 128 + c) * 1024 + wcol] = v;
      rp += v * W_read[c];
    }
  // h contribution to readout (f32 source for precision); halves split c-range
  const float* hb = h + (size_t)b * 64 * 1024 + wcol;
#pragma unroll
  for (int t = 0; t < 32; ++t) {
    int c = 32 * hl + t;
    rp += W_read[64 + c] * hb[(size_t)c * 1024];
  }
  float other = __shfl_xor(rp, 16, 32);
  rp += other;
  if (hl == 0) out_read[(size_t)b * 1024 + wcol] = rp + b_read[0];
}

// ---------------------------------------------------------------------------
extern "C" void kernel_launch(void* const* d_in, const int* in_sizes, int n_in,
                              void* d_out, int out_size, void* d_ws, size_t ws_size,
                              hipStream_t stream) {
  const float* x       = (const float*)d_in[0];
  const float* m       = (const float*)d_in[1];
  const float* u       = (const float*)d_in[2];
  const float* h       = (const float*)d_in[3];
  const float* adj     = (const float*)d_in[4];
  const float* W_in    = (const float*)d_in[5];
  const float* b_in    = (const float*)d_in[6];
  const float* W_gc    = (const float*)d_in[7];
  const float* b_gc    = (const float*)d_in[8];
  const float* W_out   = (const float*)d_in[9];
  const float* b_out   = (const float*)d_in[10];
  const float* W_read  = (const float*)d_in[11];
  const float* b_read  = (const float*)d_in[12];
  const float* prelu_w = (const float*)d_in[13];

  float* out_read = (float*)d_out;                 // [128,1,1024]
  float* out_full = out_read + 128 * 1024;         // [128,128,1024]

  char* ws = (char*)d_ws;
  __bf16* ws_xin  = (__bf16*)(ws);
  __bf16* ws_adj  = (__bf16*)(ws + 16777216u);
  __bf16* ws_hT   = (__bf16*)(ws + 20971520u);
  __bf16* ws_wgc  = (__bf16*)(ws + 37748736u);
  __bf16* ws_wout = (__bf16*)(ws + 37748736u + 16384u);

  k_prep_w  <<<1,    256, 0, stream>>>(W_gc, W_out, ws_wgc, ws_wout);
  k_prep_adj<<<2048, 256, 0, stream>>>(adj, ws_adj);
  k_prep_xin<<<512,  256, 0, stream>>>(x, m, u, h, W_in, b_in, ws_xin);
  k_prep_h  <<<2048, 256, 0, stream>>>(h, ws_hT, out_full);
  k_main    <<<2048, 128, 0, stream>>>(h, b_gc, b_out, W_read, b_read, prelu_w,
                                       ws_xin, ws_adj, ws_hT, ws_wgc, ws_wout,
                                       out_read, out_full);
}
